// TRMamba2AttnBlock_8211977470179
// MI455X (gfx1250) — matrix operand order repaired and checked by reading, not compiled, not run on hardware
//
#include <hip/hip_runtime.h>
#include <hip/hip_bf16.h>
#include <math.h>

// ---------------------------------------------------------------------------
// MI455X (gfx1250) implementation of the TRMamba2Attn block.
// All GEMMs run on v_wmma_f32_16x16x32_bf16 (f32 accum), activations converted
// f32->bf16 on the fly; softmax / norms / epilogues in f32.
// ---------------------------------------------------------------------------

typedef __attribute__((ext_vector_type(16))) __bf16 v16bf;
typedef __attribute__((ext_vector_type(8)))  __bf16 v8bf;
typedef __attribute__((ext_vector_type(8)))  float  v8f;

union FragBF { v16bf v; v8bf h[2]; };
union Pack4  { uint2 u; __bf16 b[4]; };

static __device__ __forceinline__ v8f wmma_bf16(v16bf a, v16bf b, v8f c) {
  // (neg_a, A, neg_b, B, c_mod, C, reuse_a, reuse_b)
  return __builtin_amdgcn_wmma_f32_16x16x32_bf16(false, a, false, b, (short)0, c, false, false);
}

static __device__ __forceinline__ float silu_f(float v) {
  // v * sigmoid(v) with fast v_rcp_f32 (avoids the IEEE divide expansion)
  return v * __builtin_amdgcn_rcpf(1.0f + __expf(-v));
}

#define D_MODEL 1024
#define N_HEADS 16
#define HEAD_DIM 64
#define SEQ 2048
#define BATCH 2
#define ROWS (BATCH * SEQ)          // 4096
#define D_FF 4096

// ---------------------------------------------------------------------------
// f32 -> bf16 weight conversion
// ---------------------------------------------------------------------------
__global__ __launch_bounds__(256, 2)
void cvt_bf16_kernel(const float* __restrict__ s, __bf16* __restrict__ d, int n) {
  int i = blockIdx.x * 256 + threadIdx.x;
  if (i < n) d[i] = (__bf16)s[i];
}

// ---------------------------------------------------------------------------
// Fused: x += rms(x,w1); x += rms(x,w2); h = rms(x,w3).  One block per row.
// ---------------------------------------------------------------------------
__global__ __launch_bounds__(256, 2)
void prep_kernel(const float* __restrict__ x,
                 const float* __restrict__ w1,
                 const float* __restrict__ w2,
                 const float* __restrict__ w3,
                 float* __restrict__ x1out,
                 float* __restrict__ hout) {
  __shared__ float red[256];
  const int row = blockIdx.x;
  const int t = threadIdx.x;
  const float* xr = x + (size_t)row * D_MODEL;
  float4 v = ((const float4*)xr)[t];

  float ss = v.x*v.x + v.y*v.y + v.z*v.z + v.w*v.w;
  red[t] = ss; __syncthreads();
  for (int s = 128; s > 0; s >>= 1) { if (t < s) red[t] += red[t+s]; __syncthreads(); }
  float rs = rsqrtf(red[0] * (1.0f/D_MODEL) + 1e-6f); __syncthreads();
  float4 wv = ((const float4*)w1)[t];
  v.x += v.x*rs*wv.x; v.y += v.y*rs*wv.y; v.z += v.z*rs*wv.z; v.w += v.w*rs*wv.w;

  ss = v.x*v.x + v.y*v.y + v.z*v.z + v.w*v.w;
  red[t] = ss; __syncthreads();
  for (int s = 128; s > 0; s >>= 1) { if (t < s) red[t] += red[t+s]; __syncthreads(); }
  rs = rsqrtf(red[0] * (1.0f/D_MODEL) + 1e-6f); __syncthreads();
  wv = ((const float4*)w2)[t];
  v.x += v.x*rs*wv.x; v.y += v.y*rs*wv.y; v.z += v.z*rs*wv.z; v.w += v.w*rs*wv.w;

  ss = v.x*v.x + v.y*v.y + v.z*v.z + v.w*v.w;
  red[t] = ss; __syncthreads();
  for (int s = 128; s > 0; s >>= 1) { if (t < s) red[t] += red[t+s]; __syncthreads(); }
  rs = rsqrtf(red[0] * (1.0f/D_MODEL) + 1e-6f);
  wv = ((const float4*)w3)[t];
  float4 h;
  h.x = v.x*rs*wv.x; h.y = v.y*rs*wv.y; h.z = v.z*rs*wv.z; h.w = v.w*rs*wv.w;

  ((float4*)(x1out + (size_t)row * D_MODEL))[t] = v;
  ((float4*)(hout  + (size_t)row * D_MODEL))[t] = h;
}

// Single rmsnorm: out = rms(in, w)
__global__ __launch_bounds__(256, 2)
void rms_kernel(const float* __restrict__ x,
                const float* __restrict__ w,
                float* __restrict__ out) {
  __shared__ float red[256];
  const int row = blockIdx.x;
  const int t = threadIdx.x;
  float4 v = ((const float4*)(x + (size_t)row * D_MODEL))[t];
  float ss = v.x*v.x + v.y*v.y + v.z*v.z + v.w*v.w;
  red[t] = ss; __syncthreads();
  for (int s = 128; s > 0; s >>= 1) { if (t < s) red[t] += red[t+s]; __syncthreads(); }
  float rs = rsqrtf(red[0] * (1.0f/D_MODEL) + 1e-6f);
  float4 wv = ((const float4*)w)[t];
  float4 h;
  h.x = v.x*rs*wv.x; h.y = v.y*rs*wv.y; h.z = v.z*rs*wv.z; h.w = v.w*rs*wv.w;
  ((float4*)(out + (size_t)row * D_MODEL))[t] = h;
}

// ---------------------------------------------------------------------------
// Generic bf16-WMMA GEMM:  C[M,N] = epi( A[M,K](f32) @ W[K,N](bf16) )
// BM=128, BN template (128 or 64), BK=32.  256 threads = 8 waves.
//   BN=128: wave grid 2x4, 64x32 per wave -> 8 WMMAs / k-step
//   BN= 64: wave grid 4x2, 32x32 per wave -> 4 WMMAs / k-step
// Epilogue compile-time specialized: +bias[n] | silu | *mulp[m,n] | +resid[m,n]
// Requires M%128==0, N%BN==0, K%32==0.
//
// Fragment layouts (ISA 7.12.2, wave32, 16-bit):
//   A 16x32: lane l -> row m=l&15; halves: K = 8*(l>>4)+0..7 and 16+8*(l>>4)+0..7
//   B 32x16: lane l -> col n=l&15; halves: K = 8*(l>>4)+0..7 and 16+8*(l>>4)+0..7
//   C 16x16: lane l -> col n=l&15; vgpr r -> row m = r + 8*(l>>4)
// ---------------------------------------------------------------------------
template <int BN, bool BIAS, int ACT, bool MUL, bool RES>
__global__ __launch_bounds__(256, 2)
void gemm_bf16(const float* __restrict__ A, const __bf16* __restrict__ W,
               const float* __restrict__ bias, const float* __restrict__ resid,
               const float* __restrict__ mulp, float* __restrict__ C,
               int M, int N, int K) {
  constexpr int WC = (BN == 128) ? 4 : 2;   // wave cols
  constexpr int WR = 8 / WC;                // wave rows
  constexpr int MT = 128 / (WR * 16);       // m-tiles per wave
  constexpr int NT = BN / (WC * 16);        // n-tiles per wave (=2)

  __shared__ __align__(16) __bf16 Al[128][40];  // row-major A tile (pad to 40 halfs)
  __shared__ __align__(16) __bf16 Bl[BN][40];   // transposed W tile: Bl[n][k]

  const int m0 = blockIdx.y * 128;
  const int n0 = blockIdx.x * BN;
  const int t = threadIdx.x;
  const int w = t >> 5, lane = t & 31, ln = lane & 15, hi = lane >> 4;
  const int wr = w / WC, wc = w % WC;

  const v8f vzero = {0.f,0.f,0.f,0.f,0.f,0.f,0.f,0.f};
  v8f acc[MT][NT];
#pragma unroll
  for (int a = 0; a < MT; ++a)
#pragma unroll
    for (int b = 0; b < NT; ++b) acc[a][b] = vzero;

  for (int k0 = 0; k0 < K; k0 += 32) {
    __syncthreads();
    // ---- stage A tile (128x32 f32 -> bf16) ----
    {
      const int col4 = (t & 7) * 4;
      const int rb = t >> 3;            // 0..31
#pragma unroll
      for (int p = 0; p < 4; ++p) {
        const int r = rb + p * 32;
        float4 a = *(const float4*)(A + (size_t)(m0 + r) * K + k0 + col4);
        Pack4 pk;
        pk.b[0] = (__bf16)a.x; pk.b[1] = (__bf16)a.y;
        pk.b[2] = (__bf16)a.z; pk.b[3] = (__bf16)a.w;
        *(uint2*)&Al[r][col4] = pk.u;
      }
    }
    // ---- stage W tile transposed (32 x BN bf16 -> Bl[n][k]) ----
    {
      constexpr int TPR = BN / 4;       // threads per k-row
      constexpr int RPP = 256 / TPR;    // k-rows per pass
      const int c4 = (t % TPR) * 4;
      const int kr = t / TPR;
#pragma unroll
      for (int p = 0; p < 32 / RPP; ++p) {
        const int kk = kr + p * RPP;
        Pack4 pk;
        pk.u = *(const uint2*)(W + (size_t)(k0 + kk) * N + n0 + c4);
#pragma unroll
        for (int i = 0; i < 4; ++i) Bl[c4 + i][kk] = pk.b[i];
      }
    }
    __syncthreads();

    // ---- fragments + WMMAs ----
    FragBF af[MT], bf_[NT];
#pragma unroll
    for (int a = 0; a < MT; ++a) {
      const int m = wr * (MT * 16) + a * 16 + ln;
      af[a].h[0] = *(const v8bf*)&Al[m][8 * hi];
      af[a].h[1] = *(const v8bf*)&Al[m][16 + 8 * hi];
    }
#pragma unroll
    for (int b = 0; b < NT; ++b) {
      const int n = wc * (NT * 16) + b * 16 + ln;
      bf_[b].h[0] = *(const v8bf*)&Bl[n][8 * hi];
      bf_[b].h[1] = *(const v8bf*)&Bl[n][16 + 8 * hi];
    }
#pragma unroll
    for (int a = 0; a < MT; ++a)
#pragma unroll
      for (int b = 0; b < NT; ++b)
        acc[a][b] = wmma_bf16(af[a].v, bf_[b].v, acc[a][b]);
  }

  // ---- epilogue (fully specialized, straight-line) ----
#pragma unroll
  for (int a = 0; a < MT; ++a) {
#pragma unroll
    for (int b = 0; b < NT; ++b) {
      const int n = n0 + wc * (NT * 16) + b * 16 + ln;
      float bv = 0.0f;
      if constexpr (BIAS) bv = bias[n];
#pragma unroll
      for (int r = 0; r < 8; ++r) {
        const int m = m0 + wr * (MT * 16) + a * 16 + 8 * hi + r;
        float v = acc[a][b][r] + bv;
        if constexpr (ACT == 1) v = silu_f(v);
        const size_t idx = (size_t)m * N + n;
        if constexpr (MUL) v *= mulp[idx];
        if constexpr (RES) v += resid[idx];
        C[idx] = v;
      }
    }
  }
}

// ---------------------------------------------------------------------------
// Causal flash attention, d=64. 256 threads = 8 waves; wave w owns 16 query
// rows; workgroup owns 128 queries of one (b,h). Key tiles of 64.
// ---------------------------------------------------------------------------
__global__ __launch_bounds__(256, 2)
void flash_attn(const float* __restrict__ q, const float* __restrict__ k,
                const float* __restrict__ v, float* __restrict__ ao) {
  const int qb = blockIdx.x * 128;
  const int hh = blockIdx.y;
  const int bb = blockIdx.z;
  const int t = threadIdx.x;
  const int w = t >> 5, lane = t & 31, ln = lane & 15, hi = lane >> 4;

  __shared__ __align__(16) __bf16 Kl[64][72];      // Kl[key][d]
  __shared__ __align__(16) __bf16 Vt[64][72];      // Vt[d][key]
  __shared__ __align__(16) __bf16 Pl[8][16][72];   // per-wave P strip [qrow][key]

  const float scale = 0.125f;                       // 64^-0.5
  const v8f vzero = {0.f,0.f,0.f,0.f,0.f,0.f,0.f,0.f};

  // ---- Q fragments (registers, scaled, bf16) ----
  FragBF qf[2];
  {
    const int qrow = qb + w * 16 + ln;
    const float* qp = q + (((size_t)(bb * SEQ + qrow)) * N_HEADS + hh) * HEAD_DIM;
#pragma unroll
    for (int c = 0; c < 2; ++c) {
#pragma unroll
      for (int run = 0; run < 2; ++run) {
        const int d0 = 32 * c + run * 16 + 8 * hi;
        float4 a = *(const float4*)(qp + d0);
        float4 b = *(const float4*)(qp + d0 + 4);
        v8bf hv;
        hv[0] = (__bf16)(a.x * scale); hv[1] = (__bf16)(a.y * scale);
        hv[2] = (__bf16)(a.z * scale); hv[3] = (__bf16)(a.w * scale);
        hv[4] = (__bf16)(b.x * scale); hv[5] = (__bf16)(b.y * scale);
        hv[6] = (__bf16)(b.z * scale); hv[7] = (__bf16)(b.w * scale);
        qf[c].h[run] = hv;
      }
    }
  }

  v8f o[4]; float mrow[8], lrow[8];
#pragma unroll
  for (int d = 0; d < 4; ++d) o[d] = vzero;
#pragma unroll
  for (int r = 0; r < 8; ++r) { mrow[r] = -1e30f; lrow[r] = 0.0f; }

  const int jend = qb + 127;
  for (int j0 = 0; j0 <= jend; j0 += 64) {
    __syncthreads();
    // ---- stage K tile and transposed V tile (f32 -> bf16) ----
    {
      const int rr = t >> 4;          // 0..15
      const int d0 = (t & 15) * 4;
#pragma unroll
      for (int p = 0; p < 4; ++p) {
        const int key = rr + p * 16;  // 0..63
        const size_t base = (((size_t)(bb * SEQ + j0 + key)) * N_HEADS + hh) * HEAD_DIM + d0;
        float4 kv = *(const float4*)(k + base);
        Pack4 pk;
        pk.b[0] = (__bf16)kv.x; pk.b[1] = (__bf16)kv.y;
        pk.b[2] = (__bf16)kv.z; pk.b[3] = (__bf16)kv.w;
        *(uint2*)&Kl[key][d0] = pk.u;
        float4 vv = *(const float4*)(v + base);
        Vt[d0 + 0][key] = (__bf16)vv.x;
        Vt[d0 + 1][key] = (__bf16)vv.y;
        Vt[d0 + 2][key] = (__bf16)vv.z;
        Vt[d0 + 3][key] = (__bf16)vv.w;
      }
    }
    __syncthreads();

    // ---- S = Q @ K^T  (16 x 64 per wave) ----
    v8f s4[4];
#pragma unroll
    for (int nt = 0; nt < 4; ++nt) s4[nt] = vzero;
#pragma unroll
    for (int nt = 0; nt < 4; ++nt) {
#pragma unroll
      for (int c = 0; c < 2; ++c) {
        FragBF bf_;
        bf_.h[0] = *(const v8bf*)&Kl[nt * 16 + ln][32 * c + 8 * hi];
        bf_.h[1] = *(const v8bf*)&Kl[nt * 16 + ln][32 * c + 16 + 8 * hi];
        s4[nt] = wmma_bf16(qf[c].v, bf_.v, s4[nt]);
      }
    }

    // ---- causal mask ----
#pragma unroll
    for (int nt = 0; nt < 4; ++nt) {
      const int key = j0 + nt * 16 + ln;
#pragma unroll
      for (int r = 0; r < 8; ++r) {
        const int srow = qb + w * 16 + 8 * hi + r;
        if (key > srow) s4[nt][r] = -1e30f;
      }
    }

    // ---- online softmax; write P (bf16) to per-wave LDS strip ----
#pragma unroll
    for (int r = 0; r < 8; ++r) {
      float mx = fmaxf(fmaxf(s4[0][r], s4[1][r]), fmaxf(s4[2][r], s4[3][r]));
#pragma unroll
      for (int off = 1; off < 16; off <<= 1)
        mx = fmaxf(mx, __shfl_xor(mx, off, 32));
      const float mnew = fmaxf(mrow[r], mx);
      const float corr = __expf(mrow[r] - mnew);
      mrow[r] = mnew;
      float ps = 0.0f;
#pragma unroll
      for (int nt = 0; nt < 4; ++nt) {
        const float p = __expf(s4[nt][r] - mnew);
        ps += p;
        Pl[w][8 * hi + r][nt * 16 + ln] = (__bf16)p;
      }
#pragma unroll
      for (int off = 1; off < 16; off <<= 1)
        ps += __shfl_xor(ps, off, 32);
      lrow[r] = lrow[r] * corr + ps;
#pragma unroll
      for (int d = 0; d < 4; ++d) o[d][r] *= corr;
    }

    // same-wave LDS RAW: ds ops are in-order per wave; keep compiler honest.
    asm volatile("" ::: "memory");

    // ---- O += P @ V ----
#pragma unroll
    for (int c = 0; c < 2; ++c) {
      FragBF pa;
      pa.h[0] = *(const v8bf*)&Pl[w][ln][32 * c + 8 * hi];
      pa.h[1] = *(const v8bf*)&Pl[w][ln][32 * c + 16 + 8 * hi];
#pragma unroll
      for (int dt = 0; dt < 4; ++dt) {
        FragBF vb;
        vb.h[0] = *(const v8bf*)&Vt[dt * 16 + ln][32 * c + 8 * hi];
        vb.h[1] = *(const v8bf*)&Vt[dt * 16 + ln][32 * c + 16 + 8 * hi];
        o[dt] = wmma_bf16(pa.v, vb.v, o[dt]);
      }
    }
  }

  // ---- normalize (fast rcp) + store ----
  float invl[8];
#pragma unroll
  for (int r = 0; r < 8; ++r) invl[r] = __builtin_amdgcn_rcpf(lrow[r]);
#pragma unroll
  for (int dt = 0; dt < 4; ++dt) {
#pragma unroll
    for (int r = 0; r < 8; ++r) {
      const int srow = qb + w * 16 + 8 * hi + r;
      const float val = o[dt][r] * invl[r];
      ao[(((size_t)(bb * SEQ + srow)) * N_HEADS + hh) * HEAD_DIM + dt * 16 + ln] = val;
    }
  }
}

// ---------------------------------------------------------------------------
// Host launcher
// ---------------------------------------------------------------------------
extern "C" void kernel_launch(void* const* d_in, const int* in_sizes, int n_in,
                              void* d_out, int out_size, void* d_ws, size_t ws_size,
                              hipStream_t stream) {
  (void)in_sizes; (void)n_in; (void)out_size; (void)ws_size;
  const float* x      = (const float*)d_in[0];
  // d_in[1] = attn_mask (causal) — implemented analytically.
  const float* n1w    = (const float*)d_in[2];
  const float* n2w    = (const float*)d_in[3];
  const float* n3w    = (const float*)d_in[4];
  const float* mlpnw  = (const float*)d_in[5];
  const float* wq     = (const float*)d_in[6];
  const float* wk     = (const float*)d_in[7];
  const float* wv     = (const float*)d_in[8];
  const float* wo     = (const float*)d_in[9];
  const float* qs_w1  = (const float*)d_in[10];
  const float* qs_b1  = (const float*)d_in[11];
  const float* qs_w2  = (const float*)d_in[12];
  const float* qs_b2  = (const float*)d_in[13];
  const float* ks_w1  = (const float*)d_in[14];
  const float* ks_b1  = (const float*)d_in[15];
  const float* ks_w2  = (const float*)d_in[16];
  const float* ks_b2  = (const float*)d_in[17];
  const float* fc1_w  = (const float*)d_in[18];
  const float* fc1_b  = (const float*)d_in[19];
  const float* fc2_w  = (const float*)d_in[20];
  const float* fc2_b  = (const float*)d_in[21];
  const float* gate_w = (const float*)d_in[22];
  const float* gate_b = (const float*)d_in[23];

  char* ws = (char*)d_ws;
  const size_t MB = 1ull << 20;
  // bf16 weights (persistent)
  __bf16* wqB   = (__bf16*)(ws + 0 * MB);
  __bf16* wkB   = (__bf16*)(ws + 2 * MB);
  __bf16* wvB   = (__bf16*)(ws + 4 * MB);
  __bf16* woB   = (__bf16*)(ws + 6 * MB);
  __bf16* fc1B  = (__bf16*)(ws + 8 * MB);
  __bf16* gateB = (__bf16*)(ws + 16 * MB);
  __bf16* fc2B  = (__bf16*)(ws + 24 * MB);
  __bf16* qs1B  = (__bf16*)(ws + 32 * MB);
  __bf16* qs2B  = (__bf16*)(ws + 32 * MB + 16384);
  __bf16* ks1B  = (__bf16*)(ws + 32 * MB + 32768);
  __bf16* ks2B  = (__bf16*)(ws + 32 * MB + 49152);
  // f32 activations (lifetime-based reuse; total ws footprint = 225 MB)
  float* x1 = (float*)(ws + 33  * MB);   // residual after two mamba stages
  float* h  = (float*)(ws + 49  * MB);   // rms(x, norm3)
  float* qf = (float*)(ws + 65  * MB);   // dead after q2
  float* kf = (float*)(ws + 81  * MB);   // dead after k2
  float* vf = (float*)(ws + 97  * MB);   // dead after attention
  float* t1 = (float*)(ws + 113 * MB);   // 32MB head-MLP hidden, dead after k2
  float* q2 = (float*)(ws + 145 * MB);   // dead after attention
  float* k2 = (float*)(ws + 161 * MB);   // dead after attention
  float* ao = (float*)(ws + 177 * MB);   // dead after wo-gemm
  float* xa = (float*)(ws + 193 * MB);
  float* hn = (float*)(ws + 209 * MB);
  float* g  = (float*)(ws + 65  * MB);   // 64MB, reuses qf/kf/vf/t1-head
  float* y  = (float*)(ws + 129 * MB);   // 64MB, reuses t1-tail/q2/k2/ao

  auto cvt = [&](const float* src, __bf16* dst, int n) {
    cvt_bf16_kernel<<<(n + 255) / 256, 256, 0, stream>>>(src, dst, n);
  };
  cvt(wq, wqB, D_MODEL * D_MODEL);
  cvt(wk, wkB, D_MODEL * D_MODEL);
  cvt(wv, wvB, D_MODEL * D_MODEL);
  cvt(wo, woB, D_MODEL * D_MODEL);
  cvt(qs_w1, qs1B, HEAD_DIM * 128);
  cvt(qs_w2, qs2B, 128 * HEAD_DIM);
  cvt(ks_w1, ks1B, HEAD_DIM * 128);
  cvt(ks_w2, ks2B, 128 * HEAD_DIM);
  cvt(fc1_w, fc1B, D_MODEL * D_FF);
  cvt(gate_w, gateB, D_MODEL * D_FF);
  cvt(fc2_w, fc2B, D_FF * D_MODEL);

  // mamba identity stages + attention pre-norm
  prep_kernel<<<ROWS, 256, 0, stream>>>(x, n1w, n2w, n3w, x1, h);

  const int MH = ROWS * N_HEADS;  // 65536 head-rows of 64

  // QKV projections (BN=128, plain)
  {
    dim3 g128(D_MODEL / 128, ROWS / 128);
    gemm_bf16<128, false, 0, false, false><<<g128, 256, 0, stream>>>(
        h, wqB, nullptr, nullptr, nullptr, qf, ROWS, D_MODEL, D_MODEL);
    gemm_bf16<128, false, 0, false, false><<<g128, 256, 0, stream>>>(
        h, wkB, nullptr, nullptr, nullptr, kf, ROWS, D_MODEL, D_MODEL);
    gemm_bf16<128, false, 0, false, false><<<g128, 256, 0, stream>>>(
        h, wvB, nullptr, nullptr, nullptr, vf, ROWS, D_MODEL, D_MODEL);
  }

  // per-head SiLU-MLP residual transforms
  {
    dim3 gs1(128 / 128, MH / 128);      // N=128
    dim3 gs2(64 / 64,  MH / 128);       // N=64
    gemm_bf16<128, true, 1, false, false><<<gs1, 256, 0, stream>>>(
        qf, qs1B, qs_b1, nullptr, nullptr, t1, MH, 128, HEAD_DIM);
    gemm_bf16<64, true, 0, false, true><<<gs2, 256, 0, stream>>>(
        t1, qs2B, qs_b2, qf, nullptr, q2, MH, HEAD_DIM, 128);
    gemm_bf16<128, true, 1, false, false><<<gs1, 256, 0, stream>>>(
        kf, ks1B, ks_b1, nullptr, nullptr, t1, MH, 128, HEAD_DIM);
    gemm_bf16<64, true, 0, false, true><<<gs2, 256, 0, stream>>>(
        t1, ks2B, ks_b2, kf, nullptr, k2, MH, HEAD_DIM, 128);
  }

  // causal flash attention
  dim3 agrid(SEQ / 128, N_HEADS, BATCH);
  flash_attn<<<agrid, 256, 0, stream>>>(q2, k2, vf, ao);

  // output projection + residual
  gemm_bf16<128, false, 0, false, true><<<dim3(D_MODEL / 128, ROWS / 128), 256, 0, stream>>>(
      ao, woB, nullptr, x1, nullptr, xa, ROWS, D_MODEL, D_MODEL);

  // MLP: hn = rms(xa); g = silu(hn@gate+b); y = g*(hn@fc1+b); out = xa + y@fc2 + b
  rms_kernel<<<ROWS, 256, 0, stream>>>(xa, mlpnw, hn);
  {
    dim3 gff(D_FF / 128, ROWS / 128);
    gemm_bf16<128, true, 1, false, false><<<gff, 256, 0, stream>>>(
        hn, gateB, gate_b, nullptr, nullptr, g, ROWS, D_FF, D_MODEL);
    gemm_bf16<128, true, 0, true, false><<<gff, 256, 0, stream>>>(
        hn, fc1B, fc1_b, nullptr, g, y, ROWS, D_FF, D_MODEL);
    dim3 gout(D_MODEL / 128, ROWS / 128);
    gemm_bf16<128, true, 0, false, true><<<gout, 256, 0, stream>>>(
        y, fc2B, fc2_b, xa, nullptr, (float*)d_out, ROWS, D_MODEL, D_FF);
  }
}
